// SSTv2_70798240907492
// MI455X (gfx1250) — matrix operand order repaired and checked by reading, not compiled, not run on hardware
//
#include <hip/hip_runtime.h>

typedef __attribute__((ext_vector_type(16))) _Float16 v16h;
typedef __attribute__((ext_vector_type(8)))  float    v8f;

// ---------------- problem constants ----------------
constexpr int kN    = 120000;
constexpr int kC    = 128;
constexpr int kNH   = 8;
constexpr int kHD   = 16;
constexpr int kT    = 36;
constexpr int kFILL = 24;
constexpr int kNW   = 5000;     // kN / kFILL
constexpr int kL    = 4;
constexpr int kDFF  = 256;
constexpr int kB    = 2;
constexpr int kH    = 320;
constexpr int kW    = 320;
constexpr int kTP   = 48;       // T padded to 3x16 tiles

// ---------------- CDNA5 async copy helpers ----------------
// Low 32 bits of a generic pointer to LDS == workgroup-relative LDS offset
// (ISA 10.2); that is what async VDST expects.
static __device__ __forceinline__ void async_g2l_b128(void* lds_dst,
                                                      const void* gsrc) {
  unsigned lds = (unsigned)(size_t)lds_dst;
  unsigned long long g = (unsigned long long)(size_t)gsrc;
  asm volatile("global_load_async_to_lds_b128 %0, %1, off"
               :: "v"(lds), "v"(g) : "memory");
}
static __device__ __forceinline__ void wait_async0() {
  asm volatile("s_wait_asynccnt 0" ::: "memory");
}

// ---------------- WMMA helpers (wave32, 16x16x32 f16 -> f32) ----------------
static __device__ __forceinline__ v8f wmma_ff(v16h a, v16h b, v8f c) {
  return __builtin_amdgcn_wmma_f32_16x16x32_f16(false, a, false, b, (short)0, c,
                                                false, false);
}

// A fragment 16x32 from row-major f16 source; element zero when k >= kmax.
static __device__ __forceinline__ v16h ldA(const _Float16* S, int m0, int k0,
                                           int ld, int kmax, int lane) {
  int m = lane & 15, half = lane >> 4;
  v16h a;
#pragma unroll
  for (int j = 0; j < 16; ++j) {
    int k = k0 + half * 16 + j;
    a[j] = (k < kmax) ? S[(m0 + m) * ld + k] : (_Float16)0;
  }
  return a;
}

// B fragment 32x16 where B[k][n] = W[n0+n][k0+k]  (Y = X @ W^T, W row-major [n][k])
static __device__ __forceinline__ v16h ldB_w(const _Float16* W, int n0, int k0,
                                             int ldw, int lane) {
  int n = lane & 15, kh = lane >> 4;
  v16h b;
#pragma unroll
  for (int j = 0; j < 16; ++j) b[j] = W[(n0 + n) * ldw + k0 + kh * 16 + j];
  return b;
}

// B fragment 32x16 where B[k][n] = S[k0+k][n0+n] (row-major K x N); zero if k>=kmax
static __device__ __forceinline__ v16h ldB_rm(const _Float16* S, int k0, int n0,
                                              int ld, int kmax, int lane) {
  int n = lane & 15, kh = lane >> 4;
  v16h b;
#pragma unroll
  for (int j = 0; j < 16; ++j) {
    int k = k0 + kh * 16 + j;
    b[j] = (k < kmax) ? S[k * ld + n0 + n] : (_Float16)0;
  }
  return b;
}

// B fragment 32x16 where B[k][n] = S[n0+n][c0+k] (transposed source); k local, zero if k>=kmax
static __device__ __forceinline__ v16h ldB_t(const _Float16* S, int n0, int c0,
                                             int ld, int kmax, int lane) {
  int n = lane & 15, kh = lane >> 4;
  v16h b;
#pragma unroll
  for (int j = 0; j < 16; ++j) {
    int k = kh * 16 + j;
    b[j] = (k < kmax) ? S[(n0 + n) * ld + c0 + k] : (_Float16)0;
  }
  return b;
}

// ---------------- small utility kernels ----------------
__global__ void k_copy_f(float* dst, const float* src, int n) {
  int i = blockIdx.x * 256 + threadIdx.x;
  if (i < n) dst[i] = src[i];
}
__global__ void k_zero_h(_Float16* dst, int n) {
  int i = blockIdx.x * 256 + threadIdx.x;
  if (i < n) dst[i] = (_Float16)0;
}
__global__ void k_f2h(_Float16* dst, const float* src, int n) {
  int i = blockIdx.x * 256 + threadIdx.x;
  if (i < n) dst[i] = (_Float16)src[i];
}
// conv weight rearrange: dst[conv][tap][ic][oc] = src[conv][oc][ic][ky][kx]
__global__ void k_cvt_cw(_Float16* dst, const float* src) {
  int i = blockIdx.x * 256 + threadIdx.x;
  if (i >= 2 * 9 * kC * kC) return;
  int conv = i / (9 * kC * kC), rem = i % (9 * kC * kC);
  int tap = rem / (kC * kC), rem2 = rem % (kC * kC);
  int ic = rem2 / kC, oc = rem2 % kC;
  int ky = tap / 3, kx = tap % 3;
  dst[i] = (_Float16)src[(((size_t)conv * kC + oc) * kC + ic) * 9 + ky * 3 + kx];
}
__global__ void k_scatter_feat(_Float16* feat, const float* x, const int* ind) {
  int i = blockIdx.x * 256 + threadIdx.x;   // over N*C
  if (i >= kN * kC) return;
  int row = i >> 7, c = i & (kC - 1);
  feat[ind[row] * kC + c] = (_Float16)x[i];
}
// canvas kept in f16 channels-last so conv staging is a raw async byte copy
__global__ void k_scatter_canvas(_Float16* canvas, const float* x,
                                 const int* vc) {
  int i = blockIdx.x * 256 + threadIdx.x;   // over N*C
  if (i >= kN * kC) return;
  int row = i >> 7, c = i & (kC - 1);
  int b = vc[row * 4 + 0], yy = vc[row * 4 + 2], xx = vc[row * 4 + 3];
  canvas[(((size_t)b * kH + yy) * kW + xx) * kC + c] = (_Float16)x[i];
}

// ---------------- attention: one block per window ----------------
__global__ __launch_bounds__(256) void k_attn(
    const _Float16* __restrict__ feat, const float* __restrict__ pos,
    const _Float16* __restrict__ ipw, const float* __restrict__ ipb,
    const _Float16* __restrict__ opw, const float* __restrict__ opb,
    float* __restrict__ obuf) {
  __shared__ __align__(16) _Float16 s_feat[kTP * kC];
  __shared__ __align__(16) _Float16 s_qk[kTP * kC];
  __shared__ __align__(16) _Float16 s_q[kTP * kC];
  __shared__ __align__(16) _Float16 s_k[kTP * kC];
  __shared__ __align__(16) _Float16 s_v[kTP * kC];
  __shared__ __align__(16) _Float16 s_o[kTP * kC];
  __shared__ __align__(16) _Float16 s_attn[kNH * kTP * kTP];

  const int tid = threadIdx.x, wv = tid >> 5, lane = tid & 31;
  const int win = blockIdx.x;

  // --- async copy feat tile (kT*kC f16 = 9216 B contiguous) into s_feat ---
  for (int i = tid; i < (kT * kC) / 8; i += 256)   // 576 16-byte chunks
    async_g2l_b128(&s_feat[i * 8], feat + (size_t)win * kT * kC + i * 8);
  // zero padded rows while the async engine runs
  for (int i = kT * kC + tid; i < kTP * kC; i += 256) s_feat[i] = (_Float16)0;
  // overlap: fetch this thread's pos values into registers
  float pv[18];                                     // kT*kC/256 == 18
#pragma unroll
  for (int j = 0; j < 18; ++j)
    pv[j] = pos[(size_t)win * kT * kC + tid + j * 256];
  wait_async0();
  __syncthreads();
  // qk_in = feat3d + pos
#pragma unroll
  for (int j = 0; j < 18; ++j) {
    int i = tid + j * 256;
    s_qk[i] = (_Float16)((float)s_feat[i] + pv[j]);
  }
  for (int i = kT * kC + tid; i < kTP * kC; i += 256) s_qk[i] = (_Float16)0;
  __syncthreads();

  // QKV: 3 mats x 3 mtiles x 8 ntiles = 72 tile jobs over 8 waves
  for (int job = wv; job < 72; job += 8) {
    int mat = job / 24, rem = job % 24, mt = rem / 8, nt = rem % 8;
    const _Float16* src = (mat == 2) ? s_feat : s_qk;
    v8f acc = {};
#pragma unroll
    for (int ks = 0; ks < 4; ++ks) {
      v16h a = ldA(src, mt * 16, ks * 32, kC, kC, lane);
      v16h b = ldB_w(ipw, mat * kC + nt * 16, ks * 32, kC, lane);
      acc = wmma_ff(a, b, acc);
    }
    _Float16* dst = (mat == 0) ? s_q : (mat == 1) ? s_k : s_v;
    int n = lane & 15, mh = lane >> 4;
    float bias = ipb[mat * kC + nt * 16 + n];
#pragma unroll
    for (int r = 0; r < 8; ++r)
      dst[(mt * 16 + mh * 8 + r) * kC + nt * 16 + n] = (_Float16)(acc[r] + bias);
  }
  __syncthreads();

  // scores: wave = head; q_h(48x16) @ k_h^T(16x48), K padded to 32
  {
    int h = wv;
    _Float16* at = s_attn + h * kTP * kTP;
#pragma unroll
    for (int mt = 0; mt < 3; ++mt)
#pragma unroll
      for (int nt = 0; nt < 3; ++nt) {
        v8f acc = {};
        v16h a = ldA(s_q, mt * 16, h * 16, kC, h * 16 + kHD, lane);
        v16h b = ldB_t(s_k, nt * 16, h * 16, kC, kHD, lane);
        acc = wmma_ff(a, b, acc);
        int n = lane & 15, mh = lane >> 4;
        bool maskc = (nt * 16 + n) >= kFILL;   // mask = token >= FILL
#pragma unroll
        for (int r = 0; r < 8; ++r) {
          float sv = maskc ? -60000.f : acc[r] * 0.25f;   // SCALE = HD^-0.5
          at[(mt * 16 + mh * 8 + r) * kTP + nt * 16 + n] = (_Float16)sv;
        }
      }
  }
  __syncthreads();

  // softmax in place (each lane owns whole rows)
  {
    int h = wv;
    _Float16* at = s_attn + h * kTP * kTP;
    for (int r = lane; r < kTP; r += 32) {
      if (r < kT) {
        float mx = -1e30f;
        for (int c = 0; c < kTP; ++c) mx = fmaxf(mx, (float)at[r * kTP + c]);
        float sum = 0.f;
        for (int c = 0; c < kTP; ++c) sum += __expf((float)at[r * kTP + c] - mx);
        float inv = 1.f / sum;
        for (int c = 0; c < kTP; ++c)
          at[r * kTP + c] = (_Float16)(__expf((float)at[r * kTP + c] - mx) * inv);
      } else {
        for (int c = 0; c < kTP; ++c) at[r * kTP + c] = (_Float16)0;
      }
    }
  }
  __syncthreads();

  // o_h = attn(48x48) @ v_h(48x16), K padded to 64
  {
    int h = wv;
    const _Float16* at = s_attn + h * kTP * kTP;
#pragma unroll
    for (int mt = 0; mt < 3; ++mt) {
      v8f acc = {};
#pragma unroll
      for (int ks = 0; ks < 2; ++ks) {
        v16h a = ldA(at, mt * 16, ks * 32, kTP, kTP, lane);
        v16h b = ldB_rm(s_v, ks * 32, h * 16, kC, kTP, lane);
        acc = wmma_ff(a, b, acc);
      }
      int n = lane & 15, mh = lane >> 4;
#pragma unroll
      for (int r = 0; r < 8; ++r)
        s_o[(mt * 16 + mh * 8 + r) * kC + h * 16 + n] = (_Float16)acc[r];
    }
  }
  __syncthreads();

  // out_proj: wave = ntile
  {
    int nt = wv;
#pragma unroll
    for (int mt = 0; mt < 3; ++mt) {
      v8f acc = {};
#pragma unroll
      for (int ks = 0; ks < 4; ++ks) {
        v16h a = ldA(s_o, mt * 16, ks * 32, kC, kC, lane);
        v16h b = ldB_w(opw, nt * 16, ks * 32, kC, lane);
        acc = wmma_ff(a, b, acc);
      }
      int n = lane & 15, mh = lane >> 4;
      float bias = opb[nt * 16 + n];
#pragma unroll
      for (int r = 0; r < 8; ++r) {
        int row = mt * 16 + mh * 8 + r;
        if (row < kT)
          obuf[(win * kT + row) * kC + nt * 16 + n] = acc[r] + bias;
      }
    }
  }
}

// ---------------- residual gather + LayerNorm1 ----------------
__global__ __launch_bounds__(256) void k_ln_res(
    float* __restrict__ x, const float* __restrict__ obuf,
    const int* __restrict__ ind, const float* __restrict__ w,
    const float* __restrict__ b) {
  int wv = threadIdx.x >> 5, lane = threadIdx.x & 31;
  int row = blockIdx.x * 8 + wv;
  if (row >= kN) return;
  const float* orow = obuf + (size_t)ind[row] * kC;
  float v[4], sum = 0.f;
#pragma unroll
  for (int j = 0; j < 4; ++j) {
    int c = lane + j * 32;
    v[j] = x[(size_t)row * kC + c] + orow[c];
    sum += v[j];
  }
#pragma unroll
  for (int off = 16; off; off >>= 1) sum += __shfl_xor(sum, off, 32);
  float mean = sum * (1.f / kC);
  float var = 0.f;
#pragma unroll
  for (int j = 0; j < 4; ++j) { float d = v[j] - mean; var += d * d; }
#pragma unroll
  for (int off = 16; off; off >>= 1) var += __shfl_xor(var, off, 32);
  var *= (1.f / kC);
  float inv = rsqrtf(var + 1e-5f);
#pragma unroll
  for (int j = 0; j < 4; ++j) {
    int c = lane + j * 32;
    x[(size_t)row * kC + c] = (v[j] - mean) * inv * w[c] + b[c];
  }
}

// ---------------- fused FFN (lin1 + GELU + lin2) + LayerNorm2 ----------------
__global__ __launch_bounds__(256) void k_ffn(
    float* __restrict__ x, const _Float16* __restrict__ w1,
    const float* __restrict__ b1, const _Float16* __restrict__ w2,
    const float* __restrict__ b2, const float* __restrict__ nw,
    const float* __restrict__ nb) {
  __shared__ __align__(16) _Float16 s_x[16 * kC];
  __shared__ __align__(16) _Float16 s_h[16 * kDFF];
  __shared__ float s_y[16 * kC];
  const int tid = threadIdx.x, wv = tid >> 5, lane = tid & 31;
  const int r0 = blockIdx.x * 16;

  for (int i = tid; i < 16 * kC; i += 256)
    s_x[i] = (_Float16)x[(size_t)r0 * kC + i];
  __syncthreads();

  // lin1: 16x128 @ 128x256, exact GELU
  for (int nt = wv; nt < 16; nt += 8) {
    v8f acc = {};
#pragma unroll
    for (int ks = 0; ks < 4; ++ks) {
      v16h a = ldA(s_x, 0, ks * 32, kC, kC, lane);
      v16h b = ldB_w(w1, nt * 16, ks * 32, kC, lane);
      acc = wmma_ff(a, b, acc);
    }
    int n = lane & 15, mh = lane >> 4;
    float bias = b1[nt * 16 + n];
#pragma unroll
    for (int r = 0; r < 8; ++r) {
      float vv = acc[r] + bias;
      vv = 0.5f * vv * (1.f + erff(vv * 0.70710678f));
      s_h[(mh * 8 + r) * kDFF + nt * 16 + n] = (_Float16)vv;
    }
  }
  __syncthreads();

  // lin2: 16x256 @ 256x128
  {
    int nt = wv;
    v8f acc = {};
#pragma unroll
    for (int ks = 0; ks < 8; ++ks) {
      v16h a = ldA(s_h, 0, ks * 32, kDFF, kDFF, lane);
      v16h b = ldB_w(w2, nt * 16, ks * 32, kDFF, lane);
      acc = wmma_ff(a, b, acc);
    }
    int n = lane & 15, mh = lane >> 4;
    float bias = b2[nt * 16 + n];
#pragma unroll
    for (int r = 0; r < 8; ++r)
      s_y[(mh * 8 + r) * kC + nt * 16 + n] = acc[r] + bias;
  }
  __syncthreads();

  // residual + LN2, wave handles 2 rows
  for (int rr = wv; rr < 16; rr += 8) {
    size_t row = (size_t)(r0 + rr);
    float v[4], sum = 0.f;
#pragma unroll
    for (int j = 0; j < 4; ++j) {
      int c = lane + j * 32;
      v[j] = x[row * kC + c] + s_y[rr * kC + c];
      sum += v[j];
    }
#pragma unroll
    for (int off = 16; off; off >>= 1) sum += __shfl_xor(sum, off, 32);
    float mean = sum * (1.f / kC);
    float var = 0.f;
#pragma unroll
    for (int j = 0; j < 4; ++j) { float d = v[j] - mean; var += d * d; }
#pragma unroll
    for (int off = 16; off; off >>= 1) var += __shfl_xor(var, off, 32);
    var *= (1.f / kC);
    float inv = rsqrtf(var + 1e-5f);
#pragma unroll
    for (int j = 0; j < 4; ++j) {
      int c = lane + j * 32;
      x[row * kC + c] = (v[j] - mean) * inv * nw[c] + nb[c];
    }
  }
}

// ---------------- implicit-GEMM 3x3 conv + ReLU ----------------
// in: f16 channels-last [B][H][W][C]; wt: [tap][ic][oc] f16
// out: f16 channels-last (nchw=0) or f32 NCHW (nchw=1)
__global__ __launch_bounds__(256) void k_conv(
    const _Float16* __restrict__ in, const _Float16* __restrict__ wt,
    void* __restrict__ out, int nchw) {
  __shared__ __align__(16) _Float16 s_in[16 * kC];
  const int tid = threadIdx.x, wv = tid >> 5, lane = tid & 31;
  const int bidx = blockIdx.x;
  const int xt = bidx % (kW / 16);
  const int y = (bidx / (kW / 16)) % kH;
  const int b = bidx / ((kW / 16) * kH);
  const int x0 = xt * 16;

  v8f acc = {};
  for (int tap = 0; tap < 9; ++tap) {
    int ky = tap / 3, kx = tap % 3;
    int yy = y + ky - 1;
    // stage 16 px x 128 ch (4 KB): each thread one 16-byte chunk, async
    {
      int px = tid >> 4, chunk = tid & 15;
      int xx = x0 + px + kx - 1;
      _Float16* ldst = &s_in[px * kC + chunk * 8];
      if (yy >= 0 && yy < kH && xx >= 0 && xx < kW) {
        const _Float16* g = in + (((size_t)b * kH + yy) * kW + xx) * kC + chunk * 8;
        async_g2l_b128(ldst, g);
      } else {
#pragma unroll
        for (int j = 0; j < 8; ++j) ldst[j] = (_Float16)0;
      }
    }
    wait_async0();
    __syncthreads();
#pragma unroll
    for (int ks = 0; ks < 4; ++ks) {
      v16h a = ldA(s_in, 0, ks * 32, kC, kC, lane);
      v16h bb = ldB_rm(wt + tap * kC * kC, ks * 32, wv * 16, kC, 1 << 30, lane);
      acc = wmma_ff(a, bb, acc);
    }
    __syncthreads();
  }
  int n = lane & 15, mh = lane >> 4;
#pragma unroll
  for (int r = 0; r < 8; ++r) {
    int m = mh * 8 + r;                  // pixel offset in tile
    int oc = wv * 16 + n;
    float vv = fmaxf(acc[r], 0.f);
    if (nchw)
      ((float*)out)[(((size_t)b * kC + oc) * kH + y) * kW + x0 + m] = vv;
    else
      ((_Float16*)out)[(((size_t)b * kH + y) * kW + x0 + m) * kC + oc] =
          (_Float16)vv;
  }
}

// ---------------- host launch ----------------
extern "C" void kernel_launch(void* const* d_in, const int* in_sizes, int n_in,
                              void* d_out, int out_size, void* d_ws,
                              size_t ws_size, hipStream_t stream) {
  const float* voxel_feats = (const float*)d_in[0];
  const float* pos[2] = {(const float*)d_in[1], (const float*)d_in[2]};
  const float* ipw = (const float*)d_in[3];
  const float* ipb = (const float*)d_in[4];
  const float* opw = (const float*)d_in[5];
  const float* opb = (const float*)d_in[6];
  const float* l1w = (const float*)d_in[7];
  const float* l1b = (const float*)d_in[8];
  const float* l2w = (const float*)d_in[9];
  const float* l2b = (const float*)d_in[10];
  const float* n1w = (const float*)d_in[11];
  const float* n1b = (const float*)d_in[12];
  const float* n2w = (const float*)d_in[13];
  const float* n2b = (const float*)d_in[14];
  const float* convw = (const float*)d_in[15];
  const int* vc = (const int*)d_in[16];
  const int* ind[2] = {(const int*)d_in[17], (const int*)d_in[18]};

  // workspace layout (phase B reuses phase-A attention buffers; canvas f16)
  char* w = (char*)d_ws;
  float* x = (float*)(w + 0);                          // 61,440,000 B
  _Float16* feat = (_Float16*)(w + 61440000);          // 46,080,000 B (phase A)
  float* obuf = (float*)(w + 107520000);               // 92,160,000 B (phase A)
  _Float16* canvas = (_Float16*)(w + 61440000);        // 52,428,800 B (phase B)
  _Float16* out1 = (_Float16*)(w + 113868800);         // 52,428,800 B (phase B)
  _Float16* ipw_h = (_Float16*)(w + 199680000);        // 49152 elems
  _Float16* opw_h = ipw_h + 49152;                     // 16384
  _Float16* l1w_h = opw_h + 16384;                     // 32768
  _Float16* l2w_h = l1w_h + 32768;                     // 32768
  _Float16* cw_h = l2w_h + 32768;                      // 294912

  const int NC = kN * kC;                              // 15,360,000
  k_copy_f<<<(NC + 255) / 256, 256, 0, stream>>>(x, voxel_feats, NC);

  for (int l = 0; l < kL; ++l) {
    const int* indl = ind[l & 1];
    const float* posl = pos[l & 1];
    k_f2h<<<(49152 + 255) / 256, 256, 0, stream>>>(ipw_h, ipw + l * 49152, 49152);
    k_f2h<<<(16384 + 255) / 256, 256, 0, stream>>>(opw_h, opw + l * 16384, 16384);
    k_f2h<<<(32768 + 255) / 256, 256, 0, stream>>>(l1w_h, l1w + l * 32768, 32768);
    k_f2h<<<(32768 + 255) / 256, 256, 0, stream>>>(l2w_h, l2w + l * 32768, 32768);

    const int FT = kNW * kT * kC;                      // 23,040,000
    k_zero_h<<<(FT + 255) / 256, 256, 0, stream>>>(feat, FT);
    k_scatter_feat<<<(NC + 255) / 256, 256, 0, stream>>>(feat, x, indl);

    k_attn<<<kNW, 256, 0, stream>>>(feat, posl, ipw_h, ipb + l * 384, opw_h,
                                    opb + l * kC, obuf);
    k_ln_res<<<kN / 8, 256, 0, stream>>>(x, obuf, indl, n1w + l * kC,
                                         n1b + l * kC);
    k_ffn<<<kN / 16, 256, 0, stream>>>(x, l1w_h, l1b + l * kDFF, l2w_h,
                                       l2b + l * kC, n2w + l * kC, n2b + l * kC);
  }

  const int CV = kB * kH * kW * kC;                    // 26,214,400
  k_zero_h<<<(CV + 255) / 256, 256, 0, stream>>>(canvas, CV);
  k_scatter_canvas<<<(NC + 255) / 256, 256, 0, stream>>>(canvas, x, vc);
  k_cvt_cw<<<(2 * 9 * kC * kC + 255) / 256, 256, 0, stream>>>(cw_h, convw);

  const int CB = kB * kH * (kW / 16);                  // 12,800 blocks
  k_conv<<<CB, 256, 0, stream>>>(canvas, cw_h, out1, 0);
  k_conv<<<CB, 256, 0, stream>>>(out1, cw_h + 9 * kC * kC, d_out, 1);
}
